// CustomAttention_49709951484478
// MI455X (gfx1250) — compile-verified, matrix operand order checked by reading, not run on hardware
//
#include <hip/hip_runtime.h>
#include <hip/hip_bf16.h>
#include <math.h>

typedef __attribute__((ext_vector_type(16))) __bf16 v16bf;
typedef __attribute__((ext_vector_type(8)))  __bf16 v8bf;
typedef __attribute__((ext_vector_type(8)))  float  v8f;
typedef __attribute__((ext_vector_type(4)))  int    v4i;

static constexpr int kD    = 1024;
static constexpr int k3D   = 3072;
static constexpr int kH    = 16;
static constexpr int kHD   = 64;
static constexpr int kL    = 4096;
static constexpr int kB    = 4;
static constexpr int kRows = kB * kL; // 16384

// ---- CDNA5 async global->LDS copy (ASYNCcnt) if the toolchain exposes it ----
#if defined(__gfx1250__) && __has_builtin(__builtin_amdgcn_global_load_async_to_lds_b128)
#define USE_ASYNC 1
#else
#define USE_ASYNC 0
#endif

#if USE_ASYNC
typedef __attribute__((address_space(3))) v4i as3v4i;

__device__ __forceinline__ void async_ld_b128(const void* g, void* l)
{
  // LDS generic address carries the LDS byte offset in its low 32 bits;
  // integer round-trip yields a valid AS(3) pointer without an
  // address-space-mismatched pointer cast.
  __builtin_amdgcn_global_load_async_to_lds_b128(
      (v4i*)g,
      (as3v4i*)(unsigned)(size_t)l,
      0, 0);
}
__device__ __forceinline__ void wait_async0()
{
#if __has_builtin(__builtin_amdgcn_s_wait_asynccnt)
  __builtin_amdgcn_s_wait_asynccnt(0);
#else
  asm volatile("s_wait_asynccnt 0x0" ::: "memory");
#endif
}
#endif

// ---------------------------------------------------------------------------
// Kernel 1: LayerNorm (f32 -> bf16), one block of 256 threads per row (D=1024)
// ---------------------------------------------------------------------------
__global__ __launch_bounds__(256) void k_layernorm_bf16(
    const float* __restrict__ x, const float* __restrict__ g,
    const float* __restrict__ b, __bf16* __restrict__ xn)
{
  __shared__ float s_sum[256];
  __shared__ float s_sq[256];
  const int row = blockIdx.x;
  const int tid = threadIdx.x;
  const float4 v = reinterpret_cast<const float4*>(x + (size_t)row * kD)[tid];
  s_sum[tid] = v.x + v.y + v.z + v.w;
  s_sq[tid]  = v.x * v.x + v.y * v.y + v.z * v.z + v.w * v.w;
  __syncthreads();
  for (int off = 128; off > 0; off >>= 1) {
    if (tid < off) { s_sum[tid] += s_sum[tid + off]; s_sq[tid] += s_sq[tid + off]; }
    __syncthreads();
  }
  const float mu   = s_sum[0] * (1.0f / (float)kD);
  const float var  = s_sq[0] * (1.0f / (float)kD) - mu * mu;
  const float rstd = rsqrtf(var + 1e-5f);
  const float vv[4] = {v.x, v.y, v.z, v.w};
  const int c0 = tid * 4;
  #pragma unroll
  for (int i = 0; i < 4; ++i) {
    const int c = c0 + i;
    xn[(size_t)row * kD + c] = (__bf16)(((vv[i] - mu) * rstd) * g[c] + b[c]);
  }
}

// ---------------------------------------------------------------------------
// Kernel 2: transpose + cast weight (K x N, f32, row-major) -> Bt (N x K, bf16)
// ---------------------------------------------------------------------------
__global__ __launch_bounds__(256) void k_transpose_cast(
    const float* __restrict__ w, __bf16* __restrict__ wT, int K, int N)
{
  const size_t i = (size_t)blockIdx.x * 256 + threadIdx.x; // i = n*K + k
  const int n = (int)(i / (size_t)K);
  const int k = (int)(i - (size_t)n * K);
  if (n < N) wT[i] = (__bf16)w[(size_t)k * N + n];
}

// ---------------------------------------------------------------------------
// WMMA helpers (bf16 16x16x32, wave32). LDS tiles are [rows][32] bf16,
// 64-byte rows; fragment layout per CDNA5 ISA 7.12.2:
//   lanes 0-15 : row M=lane,    K halves {0..7} and {16..23}
//   lanes 16-31: row M=lane-16, K halves {8..15} and {24..31}
// ---------------------------------------------------------------------------
__device__ __forceinline__ v16bf load_frag(const __bf16* tile, int row_block, int lane)
{
  const int half = lane >> 4;
  const __bf16* p = tile + (size_t)(row_block * 16 + (lane & 15)) * 32 + half * 8;
  const v8bf lo = *(const v8bf*)(p);
  const v8bf hi = *(const v8bf*)(p + 16);
  v16bf f;
  #pragma unroll
  for (int i = 0; i < 8; ++i) { f[i] = lo[i]; f[i + 8] = hi[i]; }
  return f;
}

__device__ __forceinline__ v8f wmma_bf16(const v16bf a, const v16bf b, v8f c)
{
  return __builtin_amdgcn_wmma_f32_16x16x32_bf16(
      false, a, false, b, (short)0, c, false, false);
}

// ===========================================================================
// Big-GEMM core: 128x128 tile, 256 threads / 8 waves, double-buffered K-loop.
// Wave w: m-blocks {2*(w&3), 2*(w&3)+1}, n-blocks {4*(w>>2) .. +3}
//   -> 8 WMMAs per K-step per wave, 1 barrier per K-step.
// The K-loop is kept as a SINGLE body (unroll 1) with an XOR-toggled LDS
// buffer offset so the accumulators stay pinned in one register tuple
// (no v_mov_b64 shuffles / hazard NOPs around the WMMAs).
// ===========================================================================
__device__ __forceinline__ void gemm_acc_128x128(
    const __bf16* __restrict__ A, int lda,
    const __bf16* __restrict__ Bt, int ldb, int K,
    __bf16* sA, __bf16* sB,          // each 2 * 128 * 32 bf16
    v8f acc[2][4], int tid)
{
  const int wave = tid >> 5;
  const int lane = tid & 31;
  const int mw = wave & 3;
  const int nw = wave >> 2;
  const int r0 = tid >> 2;          // staging rows r0 and r0+64
  const int ch = (tid & 3) * 8;     // staging 8-element chunk
  const int nsteps = K >> 5;
  const unsigned lofs = r0 * 32 + ch;   // this thread's LDS slot (elements)

  // Per-thread global staging pointers (advance by 32 elements per K-step).
  const __bf16* gA0 = &A [(size_t)r0 * lda + ch];
  const __bf16* gA1 = &A [(size_t)(r0 + 64) * lda + ch];
  const __bf16* gB0 = &Bt[(size_t)r0 * ldb + ch];
  const __bf16* gB1 = &Bt[(size_t)(r0 + 64) * ldb + ch];

  // Prologue: tile 0 -> buffer 0
#if USE_ASYNC
  async_ld_b128(gA0, &sA[lofs]);
  async_ld_b128(gA1, &sA[64 * 32 + lofs]);
  async_ld_b128(gB0, &sB[lofs]);
  async_ld_b128(gB1, &sB[64 * 32 + lofs]);
  wait_async0();
#else
  *(v8bf*)&sA[lofs]           = *(const v8bf*)gA0;
  *(v8bf*)&sA[64 * 32 + lofs] = *(const v8bf*)gA1;
  *(v8bf*)&sB[lofs]           = *(const v8bf*)gB0;
  *(v8bf*)&sB[64 * 32 + lofs] = *(const v8bf*)gB1;
#endif
  __syncthreads();

  unsigned off = 0;                       // current LDS buffer (element offset)
  #pragma unroll 1
  for (int i = 0; i < nsteps; ++i) {
    const __bf16* cA = sA + off;
    const __bf16* cB = sB + off;
    const unsigned noff = off ^ (128u * 32u);
    const bool more = (i + 1) < nsteps;
    const int kn = (i + 1) << 5;          // element offset of next K-tile

#if USE_ASYNC
    if (more) {
      async_ld_b128(gA0 + kn, &sA[noff + lofs]);
      async_ld_b128(gA1 + kn, &sA[noff + 64 * 32 + lofs]);
      async_ld_b128(gB0 + kn, &sB[noff + lofs]);
      async_ld_b128(gB1 + kn, &sB[noff + 64 * 32 + lofs]);
    }
#else
    v8bf ra0, ra1, rb0, rb1;
    if (more) {
      ra0 = *(const v8bf*)(gA0 + kn);
      ra1 = *(const v8bf*)(gA1 + kn);
      rb0 = *(const v8bf*)(gB0 + kn);
      rb1 = *(const v8bf*)(gB1 + kn);
    }
#endif

    const v16bf a0 = load_frag(cA, 2 * mw + 0, lane);
    const v16bf a1 = load_frag(cA, 2 * mw + 1, lane);
    #pragma unroll
    for (int j = 0; j < 4; ++j) {
      const v16bf bq = load_frag(cB, 4 * nw + j, lane);
      acc[0][j] = wmma_bf16(a0, bq, acc[0][j]);
      acc[1][j] = wmma_bf16(a1, bq, acc[1][j]);
    }

#if USE_ASYNC
    if (more) wait_async0();
#else
    if (more) {
      *(v8bf*)&sA[noff + lofs]           = ra0;
      *(v8bf*)&sA[noff + 64 * 32 + lofs] = ra1;
      *(v8bf*)&sB[noff + lofs]           = rb0;
      *(v8bf*)&sB[noff + 64 * 32 + lofs] = rb1;
    }
#endif
    __syncthreads();
    off = noff;
  }
}

// ---------------------------------------------------------------------------
// Small-GEMM core (kept for q@s): 64x64 tile, 128 threads / 4 waves.
// ---------------------------------------------------------------------------
__device__ __forceinline__ void stage_tile64(
    const __bf16* __restrict__ src, int ld, int k0,
    __bf16* __restrict__ dst, int tid)
{
  #pragma unroll
  for (int c = 0; c < 2; ++c) {
    const int idx = c * 128 + tid;
    const int row = idx >> 2;
    const int ch  = (idx & 3) * 8;
    *(v8bf*)&dst[row * 32 + ch] = *(const v8bf*)&src[(size_t)row * ld + k0 + ch];
  }
}

__device__ __forceinline__ void gemm_acc_64x64(
    const __bf16* __restrict__ A, int lda,
    const __bf16* __restrict__ Bt, int ldb, int K,
    __bf16* sA, __bf16* sB, v8f acc[4], int tid)
{
  const int wave = tid >> 5;
  const int lane = tid & 31;
  for (int k0 = 0; k0 < K; k0 += 32) {
    __syncthreads();
    stage_tile64(A,  lda, k0, sA, tid);
    stage_tile64(Bt, ldb, k0, sB, tid);
    __syncthreads();
    const v16bf af = load_frag(sA, wave, lane);
    #pragma unroll
    for (int j = 0; j < 4; ++j) {
      const v16bf bq = load_frag(sB, j, lane);
      acc[j] = wmma_bf16(af, bq, acc[j]);
    }
  }
  __syncthreads();
}

// ---------------------------------------------------------------------------
// Kernel 3: QKV GEMM  C(16384x3072) = xn(16384x1024) * Wqkv + b, bf16 out
// ---------------------------------------------------------------------------
__global__ __launch_bounds__(256) void k_gemm_qkv(
    const __bf16* __restrict__ xn, const __bf16* __restrict__ wT,
    const float* __restrict__ bias, __bf16* __restrict__ qkv)
{
  __shared__ __bf16 sA[2 * 128 * 32];
  __shared__ __bf16 sB[2 * 128 * 32];
  const int m0 = blockIdx.x * 128;
  const int n0 = blockIdx.y * 128;
  v8f acc[2][4] = {};
  gemm_acc_128x128(xn + (size_t)m0 * kD, kD, wT + (size_t)n0 * kD, kD, kD,
                   sA, sB, acc, threadIdx.x);
  const int lane = threadIdx.x & 31;
  const int wave = threadIdx.x >> 5;
  const int mw = wave & 3;
  const int nw = wave >> 2;
  #pragma unroll
  for (int i = 0; i < 2; ++i) {
    const int mbase = m0 + mw * 32 + i * 16 + (lane >> 4) * 8;
    #pragma unroll
    for (int j = 0; j < 4; ++j) {
      const int n = n0 + nw * 64 + j * 16 + (lane & 15);
      const float bn = bias[n];
      #pragma unroll
      for (int r = 0; r < 8; ++r)
        qkv[(size_t)(mbase + r) * k3D + n] = (__bf16)(acc[i][j][r] + bn);
    }
  }
}

// ---------------------------------------------------------------------------
// Kernel 4: per (b,h): ktv = k^T v over L, then softmax rows, store s^T bf16.
// 128 threads / 4 waves; wave w covers L-range [w*1024, w*1024+1024).
// ---------------------------------------------------------------------------
__global__ __launch_bounds__(128) void k_ktv_softmax(
    const __bf16* __restrict__ qkv, __bf16* __restrict__ sT)
{
  __shared__ float  ktv[kHD * kHD];      // 16 KB
  __shared__ __bf16 sK[4][64 * 32];      // 16 KB (wave-private tiles)
  __shared__ __bf16 sV[4][64 * 32];      // 16 KB
  const int tid  = threadIdx.x;
  const int lane = tid & 31;
  const int wave = tid >> 5;
  const int bh = blockIdx.x;
  const int b  = bh >> 4;
  const int h  = bh & 15;
  for (int i = tid; i < kHD * kHD; i += 128) ktv[i] = 0.0f;
  __syncthreads();

  const __bf16* kbase = qkv + (size_t)b * kL * k3D + kD     + h * kHD;
  const __bf16* vbase = qkv + (size_t)b * kL * k3D + 2 * kD + h * kHD;
  __bf16* aT = sK[wave];
  __bf16* bT = sV[wave];
  v8f acc[4][4] = {};

  #pragma unroll 1
  for (int step = 0; step < 32; ++step) {
    const int l0 = wave * 1024 + step * 32;
    // Wave-private staging: coalesced 16B reads along hd, transposed into
    // row-major [d][l] / [e][l] tiles (DS ops are in-order within a wave).
    #pragma unroll
    for (int c = 0; c < 8; ++c) {
      const int q = c * 32 + lane;
      const int l = q >> 3;
      const int g = (q & 7) * 8;
      const v8bf dk = *(const v8bf*)&kbase[(size_t)(l0 + l) * k3D + g];
      const v8bf dv = *(const v8bf*)&vbase[(size_t)(l0 + l) * k3D + g];
      #pragma unroll
      for (int j = 0; j < 8; ++j) {
        aT[(g + j) * 32 + l] = dk[j];
        bT[(g + j) * 32 + l] = dv[j];
      }
    }
    v16bf af[4], bf_[4];
    #pragma unroll
    for (int i = 0; i < 4; ++i) af[i]  = load_frag(aT, i, lane);
    #pragma unroll
    for (int j = 0; j < 4; ++j) bf_[j] = load_frag(bT, j, lane);
    #pragma unroll
    for (int i = 0; i < 4; ++i)
      #pragma unroll
      for (int j = 0; j < 4; ++j)
        acc[i][j] = wmma_bf16(af[i], bf_[j], acc[i][j]);
  }

  // Cross-wave reduction into shared ktv (ds_add_f32).
  #pragma unroll
  for (int i = 0; i < 4; ++i)
    #pragma unroll
    for (int j = 0; j < 4; ++j) {
      const int n = j * 16 + (lane & 15);
      #pragma unroll
      for (int r = 0; r < 8; ++r) {
        const int m = i * 16 + (lane >> 4) * 8 + r;
        atomicAdd(&ktv[m * kHD + n], acc[i][j][r]);
      }
    }
  __syncthreads();

  // Row softmax over e, store s^T[e][d] bf16 for use as B^T in q @ s.
  if (tid < kHD) {
    const int d = tid;
    float mx = -3.4e38f;
    for (int e = 0; e < kHD; ++e) mx = fmaxf(mx, ktv[d * kHD + e]);
    float sum = 0.0f;
    for (int e = 0; e < kHD; ++e) sum += __expf(ktv[d * kHD + e] - mx);
    const float inv = 1.0f / sum;
    __bf16* o = sT + (size_t)bh * kHD * kHD;
    for (int e = 0; e < kHD; ++e)
      o[e * kHD + d] = (__bf16)(__expf(ktv[d * kHD + e] - mx) * inv);
  }
}

// ---------------------------------------------------------------------------
// Kernel 5: per (b,h): attn(l,e) = q(l,d) @ s(d,e); M=4096, N=K=64, bf16 out
// ---------------------------------------------------------------------------
__global__ __launch_bounds__(128) void k_gemm_qs(
    const __bf16* __restrict__ qkv, const __bf16* __restrict__ sT,
    __bf16* __restrict__ attn)
{
  __shared__ __bf16 sA[64 * 32];
  __shared__ __bf16 sB[64 * 32];
  const int bh = blockIdx.y;
  const int b  = bh >> 4;
  const int h  = bh & 15;
  const int l0 = blockIdx.x * 64;
  const __bf16* A  = qkv + (size_t)(b * kL + l0) * k3D + h * kHD; // q block
  const __bf16* Bt = sT + (size_t)bh * kHD * kHD;                 // s^T (64x64)
  v8f acc[4] = {};
  gemm_acc_64x64(A, k3D, Bt, kHD, kHD, sA, sB, acc, threadIdx.x);
  const int lane  = threadIdx.x & 31;
  const int wave  = threadIdx.x >> 5;
  const int mbase = b * kL + l0 + wave * 16 + (lane >> 4) * 8;
  const int nbase = h * kHD + (lane & 15);
  #pragma unroll
  for (int j = 0; j < 4; ++j) {
    const int n = nbase + j * 16;
    #pragma unroll
    for (int r = 0; r < 8; ++r)
      attn[(size_t)(mbase + r) * kD + n] = (__bf16)acc[j][r];
  }
}

// ---------------------------------------------------------------------------
// Kernel 6: out-proj GEMM + bias + residual, f32 out
// ---------------------------------------------------------------------------
__global__ __launch_bounds__(256) void k_gemm_out(
    const __bf16* __restrict__ attn, const __bf16* __restrict__ wT,
    const float* __restrict__ bias, const float* __restrict__ residual,
    float* __restrict__ out)
{
  __shared__ __bf16 sA[2 * 128 * 32];
  __shared__ __bf16 sB[2 * 128 * 32];
  const int m0 = blockIdx.x * 128;
  const int n0 = blockIdx.y * 128;
  v8f acc[2][4] = {};
  gemm_acc_128x128(attn + (size_t)m0 * kD, kD, wT + (size_t)n0 * kD, kD, kD,
                   sA, sB, acc, threadIdx.x);
  const int lane = threadIdx.x & 31;
  const int wave = threadIdx.x >> 5;
  const int mw = wave & 3;
  const int nw = wave >> 2;
  #pragma unroll
  for (int i = 0; i < 2; ++i) {
    const int mbase = m0 + mw * 32 + i * 16 + (lane >> 4) * 8;
    #pragma unroll
    for (int j = 0; j < 4; ++j) {
      const int n = n0 + nw * 64 + j * 16 + (lane & 15);
      const float bn = bias[n];
      #pragma unroll
      for (int r = 0; r < 8; ++r) {
        const size_t idx = (size_t)(mbase + r) * kD + n;
        out[idx] = acc[i][j][r] + bn + residual[idx];
      }
    }
  }
}

// ---------------------------------------------------------------------------
extern "C" void kernel_launch(void* const* d_in, const int* in_sizes, int n_in,
                              void* d_out, int out_size, void* d_ws, size_t ws_size,
                              hipStream_t stream)
{
  (void)in_sizes; (void)n_in; (void)out_size; (void)ws_size;
  const float* x     = (const float*)d_in[0];
  const float* w_qkv = (const float*)d_in[1];
  const float* b_qkv = (const float*)d_in[2];
  const float* w_out = (const float*)d_in[3];
  const float* b_out = (const float*)d_in[4];
  const float* ln_g  = (const float*)d_in[5];
  const float* ln_b  = (const float*)d_in[6];
  float* out = (float*)d_out;

  // Workspace carve-up (bf16): xn | qkv | WqkvT | WoutT | sT. attn reuses xn.
  __bf16* xn    = (__bf16*)d_ws;                     // 16384*1024
  __bf16* qkv   = xn    + (size_t)kRows * kD;        // 16384*3072
  __bf16* wqkvT = qkv   + (size_t)kRows * k3D;       // 3072*1024
  __bf16* woutT = wqkvT + (size_t)k3D * kD;          // 1024*1024
  __bf16* sT    = woutT + (size_t)kD * kD;           // 64*64*64
  __bf16* attn  = xn;                                // reuse after QKV GEMM

  k_layernorm_bf16<<<kRows, 256, 0, stream>>>(x, ln_g, ln_b, xn);
  k_transpose_cast<<<(k3D * kD) / 256, 256, 0, stream>>>(w_qkv, wqkvT, kD, k3D);
  k_transpose_cast<<<(kD * kD) / 256, 256, 0, stream>>>(w_out, woutT, kD, kD);
  k_gemm_qkv<<<dim3(kRows / 128, k3D / 128), 256, 0, stream>>>(xn, wqkvT, b_qkv, qkv);
  k_ktv_softmax<<<kB * kH, 128, 0, stream>>>(qkv, sT);
  k_gemm_qs<<<dim3(kL / 64, kB * kH), 128, 0, stream>>>(qkv, sT, attn);
  k_gemm_out<<<dim3(kRows / 128, kD / 128), 256, 0, stream>>>(attn, woutT, b_out, x, out);
}
